// SPRModel_90417651515616
// MI455X (gfx1250) — matrix operand order repaired and checked by reading, not compiled, not run on hardware
//
#include <hip/hip_runtime.h>
#include <math.h>

// GNN (GCN + 2xGAT + mean-pool + FC) for gfx1250 (MI455X, wave32).
// Dense node transforms: v_wmma_f32_16x16x32_f16, K fully unrolled,
// A staged in LDS pre-swizzled to fragment order, B pre-shuffled to
// fragment order in global memory (weights tiny + reused by all blocks).
// Node-feature buffers padded to 16 rows so WMMA stores are unguarded.
// Edge gather/scatter: float4 loads + float atomics (memory-bound part).

typedef _Float16 f16;
typedef __attribute__((ext_vector_type(16))) _Float16 v16h;
typedef __attribute__((ext_vector_type(8)))  float    v8f;

#define IN_C  128
#define HID_C 64
#define OUT_C 64

// ---------- helpers ----------
__device__ __forceinline__ float atomicMaxF(float* addr, float val) {
  // monotonic float max via int/uint atomics; buffer must be initialized to -inf
  if (val >= 0.0f)
    return __int_as_float(atomicMax((int*)addr, __float_as_int(val)));
  return __uint_as_float(atomicMin((unsigned int*)addr, __float_as_uint(val)));
}

__global__ void k_fill(float* __restrict__ p, int n, float v) {
  int i = blockIdx.x * blockDim.x + threadIdx.x;
  if (i < n) p[i] = v;
}

// Build WMMA B fragments from W[K,F] (f32, row-major):
//   Bfrag[ct][kt][lane][j] = (f16) W[(kt*32 + (j>>3)*16 + (lane>>4)*8 + (j&7)) * F + ct*16 + (lane&15)]
// so each lane's v16h fragment is one contiguous 32-byte load in the GEMM.
__global__ void k_make_bfrag(const float* __restrict__ W, f16* __restrict__ Bfrag,
                             int K, int F) {
  int i = blockIdx.x * blockDim.x + threadIdx.x;
  if (i >= K * F) return;
  int j    = i & 15;
  int lane = (i >> 4) & 31;
  int rem  = i >> 9;
  int kt   = rem % (K >> 5);
  int ct   = rem / (K >> 5);
  int k = kt * 32 + (j >> 3) * 16 + (lane >> 4) * 8 + (j & 7);
  int n = lane & 15;
  Bfrag[i] = (f16)W[k * F + ct * 16 + n];
}

// ---------- WMMA GEMM: C[M,F] = A[M,K] * B[K,F] ----------
// blockDim.x = 32*(F/16); one wave per 16-column tile; block covers 16 rows.
// A tile staged in LDS in fragment order: As[kt][lane][j] where
//   row  = lane&15,  K = kt*32 + 16*(j>=8) + 8*(lane>=16) + (j&7)
// (matches the ISA 16-bit A layout), so a lane's v16h is contiguous.
// C must be padded to a multiple of 16 rows (stores are unguarded).
template<int K, int F>
__global__ void k_gemm_wmma(const float* __restrict__ A, const f16* __restrict__ Bfrag,
                            float* __restrict__ C, int M) {
  constexpr int KT = K / 32;
  __shared__ f16 As[KT][32][16];
  const int tid  = threadIdx.x;
  const int wave = tid >> 5;
  const int lane = tid & 31;
  const int rowBase = blockIdx.x * 16;
  const int half = lane >> 4;
  const int mn   = lane & 15;

  // stage 16 x K tile as float4 chunks, converting f32->f16, swizzling to
  // fragment order.  CHUNKS is a multiple of blockDim for all instantiations.
  constexpr int CHUNKS = 16 * K / 4;
  for (int idx = tid; idx < CHUNKS; idx += blockDim.x) {
    int r  = idx / (K / 4);
    int c0 = (idx % (K / 4)) * 4;
    int row = rowBase + r;
    if (row >= M) row = M - 1;                 // A unpadded: clamp (garbage rows unused)
    const float4 v = *(const float4*)(A + (unsigned)row * K + c0);
    int kt  = c0 >> 5;
    int cc  = c0 & 31;
    int lsw = r + (((cc >> 3) & 1) << 4);      // dest lane
    int j0  = (cc & 7) + ((cc >> 4) << 3);     // dest j (4 consecutive)
    f16* p = &As[kt][lsw][j0];
    p[0] = (f16)v.x; p[1] = (f16)v.y; p[2] = (f16)v.z; p[3] = (f16)v.w;
  }
  __syncthreads();

  v8f acc = {};
  const f16* bp = Bfrag + ((unsigned)(wave * KT) * 32 + lane) * 16;
#pragma unroll
  for (int kt = 0; kt < KT; ++kt) {
    v16h a = *(const v16h*)(&As[kt][lane][0]);
    v16h b = *(const v16h*)(bp + kt * 32 * 16);
    acc = __builtin_amdgcn_wmma_f32_16x16x32_f16(
        /*neg_a=*/false, a, /*neg_b=*/false, b,
        /*c_mod=*/(short)0, acc, /*reuse_a=*/false, /*reuse_b=*/false);
  }

  // C/D layout: VGPR r -> row r + 8*half, lane&15 -> column. C is padded:
  // unconditional, coalesced stores (no per-row exec juggling).
  float* cp = C + (unsigned)(rowBase + half * 8) * F + wave * 16 + mn;
#pragma unroll
  for (int r = 0; r < 8; ++r) cp[(unsigned)r * F] = acc[r];
}

// ---------- graph kernels ----------
__device__ __forceinline__ int edge_src(const long long* ei, int E, int e) {
  return (e < E) ? (int)ei[e] : (e - E);
}
__device__ __forceinline__ int edge_dst(const long long* ei, int E, int e) {
  return (e < E) ? (int)ei[E + e] : (e - E);
}

__global__ void k_degree(const long long* __restrict__ ei, int E, int Ep,
                         float* __restrict__ deg) {
  int e = blockIdx.x * blockDim.x + threadIdx.x;
  if (e >= Ep) return;
  atomicAdd(&deg[edge_dst(ei, E, e)], 1.0f);
}

__global__ void k_rsqrt(float* __restrict__ p, int n) {
  int i = blockIdx.x * blockDim.x + threadIdx.x;
  if (i >= n) return;
  float d = p[i];
  p[i] = (d > 0.0f) ? rsqrtf(fmaxf(d, 1e-12f)) : 0.0f;
}

// out[dst] += h[src] * dis[src]*dis[dst]   (F = 64, one thread per edge x 4 feats)
__global__ void k_gcn_scatter(const long long* __restrict__ ei, int E, int Ep,
                              const float* __restrict__ h, const float* __restrict__ dis,
                              float* __restrict__ out) {
  long i = (long)blockIdx.x * blockDim.x + threadIdx.x;
  if (i >= (long)Ep * 16) return;
  int c = (int)(i & 15);
  int e = (int)(i >> 4);
  int s = edge_src(ei, E, e), d = edge_dst(ei, E, e);
  float norm = dis[s] * dis[d];
  const float4 v = *(const float4*)(h + (unsigned)s * 64 + c * 4);
  float* o = out + (unsigned)d * 64 + c * 4;
  atomicAdd(o + 0, v.x * norm);
  atomicAdd(o + 1, v.y * norm);
  atomicAdd(o + 2, v.z * norm);
  atomicAdd(o + 3, v.w * norm);
}

__global__ void k_bias_act(float* __restrict__ x, const float* __restrict__ b,
                           int n, int F, int relu) {
  int i = blockIdx.x * blockDim.x + threadIdx.x;
  if (i >= n) return;
  float v = x[i] + b[i % F];
  if (relu) v = fmaxf(v, 0.0f);
  x[i] = v;
}

// al_s[n,h] = <h[n,h,:], a_src[h,:]>,  al_d likewise. thread per (node,head), F=64
template<int H>
__global__ void k_att_logits(const float* __restrict__ h, const float* __restrict__ asrc,
                             const float* __restrict__ adst, float* __restrict__ als,
                             float* __restrict__ ald, int n) {
  int i = blockIdx.x * blockDim.x + threadIdx.x;
  if (i >= n * H) return;
  int node = i / H, hd = i % H;
  const float4* row = (const float4*)(h + (unsigned)node * H * 64 + hd * 64);
  const float4* as  = (const float4*)(asrc + hd * 64);
  const float4* ad  = (const float4*)(adst + hd * 64);
  float s = 0.0f, d = 0.0f;
#pragma unroll
  for (int f = 0; f < 16; ++f) {
    float4 v = row[f], a = as[f], b = ad[f];
    s += v.x * a.x + v.y * a.y + v.z * a.z + v.w * a.w;
    d += v.x * b.x + v.y * b.y + v.z * b.z + v.w * b.w;
  }
  als[i] = s;   // index = node*H + hd
  ald[i] = d;
}

template<int H>
__global__ void k_edge_max(const long long* __restrict__ ei, int E, int Ep,
                           const float* __restrict__ als, const float* __restrict__ ald,
                           float* __restrict__ ebuf, float* __restrict__ emax) {
  int i = blockIdx.x * blockDim.x + threadIdx.x;
  if (i >= Ep * H) return;
  int e = i / H, hd = i % H;
  int s = edge_src(ei, E, e), d = edge_dst(ei, E, e);
  float v = als[s * H + hd] + ald[d * H + hd];
  v = (v > 0.0f) ? v : 0.2f * v;                  // leaky_relu(0.2)
  ebuf[i] = v;
  atomicMaxF(&emax[d * H + hd], v);
}

template<int H>
__global__ void k_edge_den(const long long* __restrict__ ei, int E, int Ep,
                           const float* __restrict__ ebuf, const float* __restrict__ emax,
                           float* __restrict__ den) {
  int i = blockIdx.x * blockDim.x + threadIdx.x;
  if (i >= Ep * H) return;
  int e = i / H, hd = i % H;
  int d = edge_dst(ei, E, e);
  atomicAdd(&den[d * H + hd], __expf(ebuf[i] - emax[d * H + hd]));
}

// out[dst,h,:] += alpha * h[src,h,:]  (F=64, thread per edge x head x 4 feats)
template<int H>
__global__ void k_edge_scatter(const long long* __restrict__ ei, int E, int Ep,
                               const float* __restrict__ h, const float* __restrict__ ebuf,
                               const float* __restrict__ emax, const float* __restrict__ den,
                               float* __restrict__ out) {
  long i = (long)blockIdx.x * blockDim.x + threadIdx.x;
  if (i >= (long)Ep * H * 16) return;
  int c = (int)(i & 15);
  long t = i >> 4;
  int hd = (int)(t % H);
  int e  = (int)(t / H);
  int s = edge_src(ei, E, e), d = edge_dst(ei, E, e);
  float alpha = __expf(ebuf[e * H + hd] - emax[d * H + hd]) / den[d * H + hd];
  const float4 v = *(const float4*)(h + (unsigned)s * H * 64 + hd * 64 + c * 4);
  float* o = out + (unsigned)d * H * 64 + hd * 64 + c * 4;
  atomicAdd(o + 0, v.x * alpha);
  atomicAdd(o + 1, v.y * alpha);
  atomicAdd(o + 2, v.z * alpha);
  atomicAdd(o + 3, v.w * alpha);
}

__global__ void k_pool(const long long* __restrict__ batch, const float* __restrict__ h,
                       float* __restrict__ sums, float* __restrict__ cnts, int n) {
  long i = (long)blockIdx.x * blockDim.x + threadIdx.x;
  if (i >= (long)n * 16) return;
  int c = (int)(i & 15);
  int node = (int)(i >> 4);
  int g = (int)batch[node];
  const float4 v = *(const float4*)(h + (unsigned)node * 64 + c * 4);
  float* o = sums + (unsigned)g * 64 + c * 4;
  atomicAdd(o + 0, v.x);
  atomicAdd(o + 1, v.y);
  atomicAdd(o + 2, v.z);
  atomicAdd(o + 3, v.w);
  if (c == 0) atomicAdd(&cnts[g], 1.0f);
}

__global__ void k_head(const float* __restrict__ sums, const float* __restrict__ cnts,
                       const float* __restrict__ Wfc, const float* __restrict__ bfc,
                       float* __restrict__ out, int G) {
  int g = blockIdx.x * blockDim.x + threadIdx.x;
  if (g >= G) return;
  float inv = 1.0f / fmaxf(cnts[g], 1.0f);
  float l0 = bfc[0], l1 = bfc[1];
  for (int f = 0; f < 64; ++f) {
    float p = sums[(unsigned)g * 64 + f] * inv;
    l0 += p * Wfc[f * 2 + 0];
    l1 += p * Wfc[f * 2 + 1];
  }
  float m = fmaxf(l0, l1);
  float lse = m + __logf(__expf(l0 - m) + __expf(l1 - m));
  out[g * 2 + 0] = l0 - lse;
  out[g * 2 + 1] = l1 - lse;
}

// ---------- launcher ----------
extern "C" void kernel_launch(void* const* d_in, const int* in_sizes, int n_in,
                              void* d_out, int out_size, void* d_ws, size_t ws_size,
                              hipStream_t stream) {
  const float*     x   = (const float*)d_in[0];
  const long long* ei  = (const long long*)d_in[1];
  const long long* bat = (const long long*)d_in[2];
  const float* W1  = (const float*)d_in[3];
  const float* b1  = (const float*)d_in[4];
  const float* W2  = (const float*)d_in[5];
  const float* as2 = (const float*)d_in[6];
  const float* ad2 = (const float*)d_in[7];
  const float* b2  = (const float*)d_in[8];
  const float* W3  = (const float*)d_in[9];
  const float* as3 = (const float*)d_in[10];
  const float* ad3 = (const float*)d_in[11];
  const float* b3  = (const float*)d_in[12];
  const float* Wfc = (const float*)d_in[13];
  const float* bfc = (const float*)d_in[14];
  float* out = (float*)d_out;

  const int N  = in_sizes[0] / IN_C;
  const int E  = in_sizes[1] / 2;
  const int Ep = E + N;                 // with self loops
  const int G  = out_size / 2;
  const int Np = (N + 15) & ~15;        // row padding for unguarded WMMA stores

  // workspace layout (256B aligned)
  char* ws = (char*)d_ws;
  size_t off = 0;
  auto alloc = [&](size_t bytes) {
    off = (off + 255) & ~(size_t)255;
    size_t o = off; off += bytes; return o;
  };
  float* h1   = (float*)(ws + alloc((size_t)Np * 64 * 4));   // GCN gemm out / reused for h3
  float* h1b  = (float*)(ws + alloc((size_t)Np * 64 * 4));   // GCN aggregated / reused for h3o
  float* h2   = (float*)(ws + alloc((size_t)Np * 128 * 4));  // GAT1 gemm out
  float* h2o  = (float*)(ws + alloc((size_t)Np * 128 * 4));  // GAT1 aggregated
  float* dis  = (float*)(ws + alloc((size_t)N * 4));
  float* als  = (float*)(ws + alloc((size_t)N * 2 * 4));
  float* ald  = (float*)(ws + alloc((size_t)N * 2 * 4));
  float* ebuf = (float*)(ws + alloc((size_t)Ep * 2 * 4));
  float* emax = (float*)(ws + alloc((size_t)N * 2 * 4));
  float* den  = (float*)(ws + alloc((size_t)N * 2 * 4));
  float* sums = (float*)(ws + alloc((size_t)G * 64 * 4));
  float* cnts = (float*)(ws + alloc((size_t)G * 4));
  f16* W1f = (f16*)(ws + alloc((size_t)IN_C * HID_C * 2));   // B fragments
  f16* W2f = (f16*)(ws + alloc((size_t)HID_C * 128 * 2));
  f16* W3f = (f16*)(ws + alloc((size_t)128 * OUT_C * 2));
  float* h3  = h1;    // safe reuse: h1 dead after GCN scatter
  float* h3o = h1b;   // safe reuse: h1b dead after GAT1 gemm

  auto nb = [](long n, int b) { return (unsigned)((n + b - 1) / b); };
  auto fill = [&](float* p, long n, float v) {
    k_fill<<<nb(n, 256), 256, 0, stream>>>(p, (int)n, v);
  };
  const float NEG_INF = -__builtin_huge_valf();

  // weight shuffle f32 -> f16 WMMA B-fragment order
  k_make_bfrag<<<nb(IN_C * HID_C, 256), 256, 0, stream>>>(W1, W1f, IN_C, HID_C);
  k_make_bfrag<<<nb(HID_C * 128, 256), 256, 0, stream>>>(W2, W2f, HID_C, 128);
  k_make_bfrag<<<nb(128 * OUT_C, 256), 256, 0, stream>>>(W3, W3f, 128, OUT_C);

  // accumulator init (workspace is poisoned; must init every call)
  fill(dis, N, 0.0f);
  fill(h1b, (long)N * 64, 0.0f);
  fill(h2o, (long)N * 128, 0.0f);
  fill(emax, (long)N * 2, NEG_INF);
  fill(den, (long)N * 2, 0.0f);
  fill(sums, (long)G * 64, 0.0f);
  fill(cnts, G, 0.0f);

  const unsigned mtiles = (unsigned)(Np / 16);

  // --- layer 1: GCNConv(128->64) ---
  k_gemm_wmma<IN_C, HID_C><<<mtiles, 32 * (HID_C / 16), 0, stream>>>(x, W1f, h1, N);
  k_degree<<<nb(Ep, 256), 256, 0, stream>>>(ei, E, Ep, dis);
  k_rsqrt<<<nb(N, 256), 256, 0, stream>>>(dis, N);
  k_gcn_scatter<<<nb((long)Ep * 16, 256), 256, 0, stream>>>(ei, E, Ep, h1, dis, h1b);
  k_bias_act<<<nb((long)N * 64, 256), 256, 0, stream>>>(h1b, b1, N * 64, 64, 1);

  // --- layer 2: GATConv(64 -> 64 x 2 heads, concat) ---
  k_gemm_wmma<HID_C, 128><<<mtiles, 32 * (128 / 16), 0, stream>>>(h1b, W2f, h2, N);
  k_att_logits<2><<<nb((long)N * 2, 256), 256, 0, stream>>>(h2, as2, ad2, als, ald, N);
  k_edge_max<2><<<nb((long)Ep * 2, 256), 256, 0, stream>>>(ei, E, Ep, als, ald, ebuf, emax);
  k_edge_den<2><<<nb((long)Ep * 2, 256), 256, 0, stream>>>(ei, E, Ep, ebuf, emax, den);
  k_edge_scatter<2><<<nb((long)Ep * 2 * 16, 256), 256, 0, stream>>>(ei, E, Ep, h2, ebuf, emax, den, h2o);
  k_bias_act<<<nb((long)N * 128, 256), 256, 0, stream>>>(h2o, b2, N * 128, 128, 1);

  // --- layer 3: GATConv(128 -> 64, 1 head) ---
  k_gemm_wmma<128, OUT_C><<<mtiles, 32 * (OUT_C / 16), 0, stream>>>(h2o, W3f, h3, N);
  k_att_logits<1><<<nb(N, 256), 256, 0, stream>>>(h3, as3, ad3, als, ald, N);
  fill(emax, N, NEG_INF);
  fill(den, N, 0.0f);
  fill(h3o, (long)N * 64, 0.0f);
  k_edge_max<1><<<nb(Ep, 256), 256, 0, stream>>>(ei, E, Ep, als, ald, ebuf, emax);
  k_edge_den<1><<<nb(Ep, 256), 256, 0, stream>>>(ei, E, Ep, ebuf, emax, den);
  k_edge_scatter<1><<<nb((long)Ep * 16, 256), 256, 0, stream>>>(ei, E, Ep, h3, ebuf, emax, den, h3o);
  k_bias_act<<<nb((long)N * 64, 256), 256, 0, stream>>>(h3o, b3, N * 64, 64, 0);

  // --- pool + head ---
  k_pool<<<nb((long)N * 16, 256), 256, 0, stream>>>(bat, h3o, sums, cnts, N);
  k_head<<<nb(G, 256), 256, 0, stream>>>(sums, cnts, Wfc, bfc, out, G);
}
